// Dodeform_22316650070605
// MI455X (gfx1250) — compile-verified, matrix operand order checked by reading, not run on hardware
//
#include <hip/hip_runtime.h>
#include <hip/hip_bf16.h>

typedef __attribute__((ext_vector_type(16))) __bf16 v16bf;
typedef __attribute__((ext_vector_type(8)))  float  v8f;

// Problem constants (B,C,H,W = 4,256,64,64; 3x3 kernel, 9 points)
#define CCH   256
#define HH    64
#define WW    64
#define NPT   9
#define KTOT  2304          // 9 * 256, n-major: k = n*256 + c
#define KSTEP 32
#define NSTEP 72            // KTOT / KSTEP
#define LDSTR 40            // LDS row stride in bf16 (80B: 16B aligned, conflict-free)

// ---------- weight f32 -> bf16 convert + permute to n-major K ----------
// wt[o][n*256 + c] = w[o][c*9 + n]
__global__ void wconv_kernel(const float* __restrict__ w,
                             __bf16* __restrict__ wt, int ntot) {
    int i = blockIdx.x * 256 + threadIdx.x;
    if (i >= ntot) return;
    int o = i / KTOT;
    int r = i - o * KTOT;
    int n = r >> 8;          // r / 256
    int c = r & 255;         // r % 256
    wt[i] = (__bf16)w[(size_t)o * KTOT + c * NPT + n];
}

// ---------- fused deformable-conv GEMM ----------
__global__ __launch_bounds__(256)
void deform_wmma_kernel(const float* __restrict__ feature,
                        const float* __restrict__ offset,
                        const __bf16* __restrict__ wt,   // [256][2304] bf16, n-major K
                        float* __restrict__ out)
{
    __shared__ float sg[4][576];                    // bilinear weights (pixel*9+n, corner)
    __shared__ int   si[4][576];                    // gather indices into raw 64x64 plane
    __shared__ __attribute__((aligned(16))) __bf16 Asmp[2][64  * LDSTR]; // sampled acts
    __shared__ __attribute__((aligned(16))) __bf16 Bwgt[2][256 * LDSTR]; // weights

    const int t      = threadIdx.x;
    const int lane   = t & 31;
    const int wv     = t >> 5;
    const int b      = blockIdx.x >> 6;   // batch
    const int h      = blockIdx.x & 63;   // row
    const int lane15 = lane & 15;
    const int khalf  = lane >> 4;

    // ---- sampling geometry: shared across all 256 channels ----
    for (int e = t; e < 576; e += 256) {
        int p = e / 9;                    // pixel (w coordinate)
        int n = e - p * 9;                // kernel point
        float ox = offset[(((size_t)b * 18 + n)     * HH + h) * WW + p];
        float oy = offset[(((size_t)b * 18 + 9 + n) * HH + h) * WW + p];
        float px = (float)(h + 1 + n / 3 - 1) + ox;   // p0 + p_n + off (padded coords)
        float py = (float)(p + 1 + n % 3 - 1) + oy;
        float flx = floorf(px), fly = floorf(py);
        float qltx = fminf(fmaxf(flx,       0.f), 65.f);
        float qlty = fminf(fmaxf(fly,       0.f), 65.f);
        float qrbx = fminf(fmaxf(flx + 1.f, 0.f), 65.f);
        float qrby = fminf(fmaxf(fly + 1.f, 0.f), 65.f);
        px = fminf(fmaxf(px, 0.f), 65.f);
        py = fminf(fmaxf(py, 0.f), 65.f);
        float dltx = 1.f + qltx - px, dlty = 1.f + qlty - py;
        float drbx = 1.f - qrbx + px, drby = 1.f - qrby + py;
        float g [4] = { dltx * dlty, drbx * drby, dltx * drby, drbx * dlty };
        float qx[4] = { qltx, qrbx, qltx, qrbx };
        float qy[4] = { qlty, qrby, qrby, qlty };
        #pragma unroll
        for (int c4 = 0; c4 < 4; ++c4) {
            int ix = (int)qx[c4], iy = (int)qy[c4];
            bool ok = (ix >= 1) && (ix <= 64) && (iy >= 1) && (iy <= 64);
            si[c4][e] = ok ? ((ix - 1) * WW + (iy - 1)) : 0;   // raw-plane index
            sg[c4][e] = ok ? g[c4] : 0.f;                      // padded zeros -> weight 0
        }
    }

    const float* fplane0 = feature + (size_t)b * CCH * (HH * WW);
    const int    obase   = wv * 32;
    const int    pthr    = t >> 2;          // this thread's pixel for slab fill
    const int    j0      = (t & 3) * 8;     // this thread's k-local chunk

    // ---- slab fill: K-step has uniform kernel-point n and contiguous channels ----
    auto fill = [&](int bufi, int kc) {
        const __bf16* wrow = wt + (size_t)t * KTOT + kc;
        float4 w0 = ((const float4*)wrow)[0];
        float4 w1 = ((const float4*)wrow)[1];
        float4 w2 = ((const float4*)wrow)[2];
        float4 w3 = ((const float4*)wrow)[3];
        if (kc + KSTEP < KTOT) __builtin_prefetch(wrow + KSTEP, 0, 3);

        int n     = kc >> 8;                 // uniform over the step
        int cbase = (kc & 255) + j0;         // 8 consecutive channels per thread
        int e     = pthr * 9 + n;
        float g0 = sg[0][e], g1 = sg[1][e], g2 = sg[2][e], g3 = sg[3][e];
        int   i0 = si[0][e], i1 = si[1][e], i2 = si[2][e], i3 = si[3][e];
        union { __bf16 hh[8]; float4 f; } pk;
        #pragma unroll
        for (int u = 0; u < 8; ++u) {
            const float* pl = fplane0 + (cbase + u) * (HH * WW);
            pk.hh[u] = (__bf16)(g0 * pl[i0] + g1 * pl[i1] + g2 * pl[i2] + g3 * pl[i3]);
        }
        float4* wdst = (float4*)&Bwgt[bufi][t * LDSTR];
        wdst[0] = w0; wdst[1] = w1; wdst[2] = w2; wdst[3] = w3;
        *(float4*)&Asmp[bufi][pthr * LDSTR + j0] = pk.f;
    };

    v8f acc[8] = {};   // 2 M-tiles (o) x 4 N-tiles (pixels), 16x16 f32 each

    __syncthreads();           // geometry ready
    fill(0, 0);                // prologue slab
    __syncthreads();

    union FU { v16bf v; float4 f[2]; };
    #pragma unroll 2
    for (int it = 0; it < NSTEP; ++it) {
        const int cur = it & 1;
        // kick off next slab first: global gather latency overlaps the WMMAs below
        if (it + 1 < NSTEP) fill(cur ^ 1, (it + 1) * KSTEP);

        const __bf16* Ab = Asmp[cur];
        const __bf16* Bb = Bwgt[cur];
        // fragments (16-bit layout: lanes 0-15 K=0-7,16-23; lanes 16-31 K=8-15,24-31)
        FU a0, a1, b0, b1, b2, b3;
        const __bf16* r0 = &Bb[(obase +      lane15) * LDSTR];
        const __bf16* r1 = &Bb[(obase + 16 + lane15) * LDSTR];
        a0.f[0] = *(const float4*)(r0 + khalf * 8);
        a0.f[1] = *(const float4*)(r0 + 16 + khalf * 8);
        a1.f[0] = *(const float4*)(r1 + khalf * 8);
        a1.f[1] = *(const float4*)(r1 + 16 + khalf * 8);
        const __bf16* p0 = &Ab[(      lane15) * LDSTR];
        const __bf16* p1 = &Ab[(16  + lane15) * LDSTR];
        const __bf16* p2 = &Ab[(32  + lane15) * LDSTR];
        const __bf16* p3 = &Ab[(48  + lane15) * LDSTR];
        b0.f[0] = *(const float4*)(p0 + khalf * 8);
        b0.f[1] = *(const float4*)(p0 + 16 + khalf * 8);
        b1.f[0] = *(const float4*)(p1 + khalf * 8);
        b1.f[1] = *(const float4*)(p1 + 16 + khalf * 8);
        b2.f[0] = *(const float4*)(p2 + khalf * 8);
        b2.f[1] = *(const float4*)(p2 + 16 + khalf * 8);
        b3.f[0] = *(const float4*)(p3 + khalf * 8);
        b3.f[1] = *(const float4*)(p3 + 16 + khalf * 8);

        acc[0] = __builtin_amdgcn_wmma_f32_16x16x32_bf16(false, a0.v, false, b0.v, (short)0, acc[0], false, false);
        acc[1] = __builtin_amdgcn_wmma_f32_16x16x32_bf16(false, a0.v, false, b1.v, (short)0, acc[1], false, false);
        acc[2] = __builtin_amdgcn_wmma_f32_16x16x32_bf16(false, a0.v, false, b2.v, (short)0, acc[2], false, false);
        acc[3] = __builtin_amdgcn_wmma_f32_16x16x32_bf16(false, a0.v, false, b3.v, (short)0, acc[3], false, false);
        acc[4] = __builtin_amdgcn_wmma_f32_16x16x32_bf16(false, a1.v, false, b0.v, (short)0, acc[4], false, false);
        acc[5] = __builtin_amdgcn_wmma_f32_16x16x32_bf16(false, a1.v, false, b1.v, (short)0, acc[5], false, false);
        acc[6] = __builtin_amdgcn_wmma_f32_16x16x32_bf16(false, a1.v, false, b2.v, (short)0, acc[6], false, false);
        acc[7] = __builtin_amdgcn_wmma_f32_16x16x32_bf16(false, a1.v, false, b3.v, (short)0, acc[7], false, false);

        __syncthreads();   // next buffer fully written; current buffer free to reuse
    }

    // ---- epilogue: relu(relu(conv) + feature); C/D layout: lane->pixel, vgpr->o ----
    const int mhalf = khalf * 8;
    #pragma unroll
    for (int mt = 0; mt < 2; ++mt) {
        #pragma unroll
        for (int nt = 0; nt < 4; ++nt) {
            v8f a = acc[mt * 4 + nt];
            int p = nt * 16 + lane15;
            #pragma unroll
            for (int r = 0; r < 8; ++r) {
                int o = obase + mt * 16 + mhalf + r;
                size_t idx = ((((size_t)b * CCH + o) * HH) + h) * WW + p;
                float v = fmaxf(a[r], 0.f) + feature[idx];
                out[idx] = fmaxf(v, 0.f);
            }
        }
    }
}

extern "C" void kernel_launch(void* const* d_in, const int* in_sizes, int n_in,
                              void* d_out, int out_size, void* d_ws, size_t ws_size,
                              hipStream_t stream) {
    const float* feature = (const float*)d_in[0];  // (4,256,64,64) f32
    const float* offset  = (const float*)d_in[1];  // (4,18,64,64)  f32
    const float* weight  = (const float*)d_in[2];  // (256,256,3,3) f32
    float* out = (float*)d_out;                    // (4,256,64,64) f32

    __bf16* wt = (__bf16*)d_ws;                    // 256*2304 bf16 = 1.18 MB
    const int nw = in_sizes[2];                    // 589824
    wconv_kernel<<<(nw + 255) / 256, 256, 0, stream>>>(weight, wt, nw);

    // one block per (batch, row): 4*64 = 256 blocks of 256 threads (8 wave32)
    deform_wmma_kernel<<<4 * HH, 256, 0, stream>>>(feature, offset, wt, out);
}
